// CPVTV2_10660108828694
// MI455X (gfx1250) — matrix operand order repaired and checked by reading, not compiled, not run on hardware
//
#include <hip/hip_runtime.h>
#include <math.h>

// ---------------- CDNA5 / gfx1250 types ----------------
typedef __bf16 bf16_t;
typedef __attribute__((ext_vector_type(16))) __bf16 v16bf;
typedef __attribute__((ext_vector_type(8)))  __bf16 v8bf;
typedef __attribute__((ext_vector_type(8)))  float  v8f;
typedef __attribute__((ext_vector_type(4)))  unsigned int v4u;
typedef __attribute__((ext_vector_type(8)))  int v8i_t;
typedef __attribute__((ext_vector_type(4)))  int v4i_t;

#if defined(__has_builtin)
#if __has_builtin(__builtin_amdgcn_tensor_load_to_lds) && __has_builtin(__builtin_amdgcn_s_wait_tensorcnt)
#define USE_TDM 1
#else
#define USE_TDM 0
#endif
#else
#define USE_TDM 0
#endif

#define BB   16
#define NN   1024
#define CC   512
#define HIDD 2048
#define CAn  128
#define MTOT (BB * NN)

// ---------------- block reduction helpers (blockDim.x == 256) ----------------
__device__ __forceinline__ float blockReduceSum(float v, float* red) {
  int tid = threadIdx.x;
  red[tid] = v; __syncthreads();
  for (int s = 128; s > 0; s >>= 1) { if (tid < s) red[tid] += red[tid + s]; __syncthreads(); }
  float r = red[0]; __syncthreads();
  return r;
}
__device__ __forceinline__ float blockReduceMax(float v, float* red) {
  int tid = threadIdx.x;
  red[tid] = v; __syncthreads();
  for (int s = 128; s > 0; s >>= 1) { if (tid < s) red[tid] = fmaxf(red[tid], red[tid + s]); __syncthreads(); }
  float r = red[0]; __syncthreads();
  return r;
}

// ---------------- fp32 -> bf16 conversion ----------------
__global__ __launch_bounds__(256)
void f2b_kernel(const float* __restrict__ in, bf16_t* __restrict__ out, int n) {
  int i = blockIdx.x * 256 + threadIdx.x;
  if (i < n) out[i] = (bf16_t)in[i];
}

// ---------------- LayerNorm over C=512, one block per token ----------------
__global__ __launch_bounds__(256)
void layernorm_kernel(const float* __restrict__ x, const float* __restrict__ g,
                      const float* __restrict__ beta, float* __restrict__ outF,
                      bf16_t* __restrict__ outB) {
  __shared__ float red[256];
  const int t = blockIdx.x;
  const int tid = threadIdx.x;
  const float* row = x + (size_t)t * CC;
  float v0 = row[tid], v1 = row[tid + 256];
  float mean = blockReduceSum(v0 + v1, red) * (1.0f / CC);
  float d0 = v0 - mean, d1 = v1 - mean;
  float var = blockReduceSum(d0 * d0 + d1 * d1, red) * (1.0f / CC);
  float rstd = rsqrtf(var + 1e-5f);
  float o0 = d0 * rstd * g[tid]       + beta[tid];
  float o1 = d1 * rstd * g[tid + 256] + beta[tid + 256];
  size_t base = (size_t)t * CC;
  if (outF) { outF[base + tid] = o0; outF[base + tid + 256] = o1; }
  outB[base + tid] = (bf16_t)o0; outB[base + tid + 256] = (bf16_t)o1;
}

// ---------------- WMMA bf16 GEMM: out[M,Ncols] = A[M,K] * W[Ncols,K]^T ----------------
// Block = 8 waves = 256(M) x 64(N) tile; each wave computes 32x64 (2 A frags x 4 B frags
// = 8 back-to-back WMMAs per K-step). B tile (64 cols x 32 K) staged to LDS by the
// Tensor Data Mover (double buffered), shared by all 8 waves. A fragments are
// register-prefetched one K-step ahead; all 4 B fragments are loaded from LDS before
// the WMMA burst so dscnt waits overlap compute.
// mode 0: store f32           mode 1: f32 = acc + bias + resid
// mode 2: bf16 = gelu(acc+b)  mode 3: f32 = acc + bias + resid
__global__ __launch_bounds__(256)
void wmma_gemm_kernel(const bf16_t* __restrict__ A, const bf16_t* __restrict__ W,
                      float* __restrict__ outF, bf16_t* __restrict__ outB,
                      const float* __restrict__ bias, const float* __restrict__ resid,
                      int M, int Ncols, int K, int mode) {
#if USE_TDM
  __shared__ bf16_t sB[2][64 * 32];          // 2 x 4KB double buffer
#endif
  const int lane  = threadIdx.x & 31;
  const int wave  = threadIdx.x >> 5;
  const int tilesN = Ncols >> 6;             // 64-wide N tiles
  const int mBlock = blockIdx.x / tilesN;
  const int nTile  = blockIdx.x - mBlock * tilesN;
  const int r16   = lane & 15;
  const int half  = lane >> 4;
  const int hiKA  = half * 8;                // A frag: lanes16-31 take K 8..15 / 24..31
  const int hiKB  = half * 8;                // B frag: lanes16-31 take K 8..15 / 24..31
  const int hiM   = half * 8;                // D: lanes16-31 hold M rows 8..15
  const int mBase = mBlock * 256 + wave * 32;
  const int nBase = nTile * 64;
  const int ksteps = K >> 5;
  if (mBase >= M) return;

#if USE_TDM
  // ---- TDM descriptor issue: load 64 rows x 32 K bf16 of W into LDS buffer ----
  auto tdm_issue = [&](int kt, int bufIdx) {
    unsigned long long ga = (unsigned long long)(const void*)(W + (size_t)nBase * K + kt);
    unsigned int ldsOff = (unsigned int)(size_t)(void*)(&sB[bufIdx][0]);
    v4u g0;
    g0[0] = 1u;                                                  // count = 1, no gather
    g0[1] = ldsOff;                                              // lds_addr (bytes)
    g0[2] = (unsigned int)(ga & 0xffffffffull);                  // global_addr[31:0]
    g0[3] = (unsigned int)((ga >> 32) & 0x1ffffffull) | (2u << 30); // addr[56:32] | type=2
    v8i_t g1;
    g1[0] = (1 << 16);                                           // data_size = 2 bytes
    g1[1] = (int)((unsigned)(K & 0xffff) << 16);                 // tensor_dim0[15:0]
    g1[2] = (int)(((unsigned)K >> 16) | (64u << 16));            // dim0[31:16] | tensor_dim1 lo
    g1[3] = (int)(32u << 16);                                    // dim1 hi(0) | tile_dim0 = 32
    g1[4] = 64;                                                  // tile_dim1 = 64, tile_dim2 = 0
    g1[5] = K;                                                   // tensor_dim0_stride[31:0]
    g1[6] = 0;                                                   // stride0 hi | stride1 lo
    g1[7] = 0;                                                   // stride1 hi
    v4i_t gz4; gz4[0] = 0; gz4[1] = 0; gz4[2] = 0; gz4[3] = 0;   // groups 2/3 unused (2D tile)
    v8i_t gz8;
#pragma unroll
    for (int i = 0; i < 8; ++i) gz8[i] = 0;                      // extra group (clang-23 form)
    __builtin_amdgcn_tensor_load_to_lds(g0, g1, gz4, gz4, gz8, 0);
  };
#endif

  v8f acc[2][4];
#pragma unroll
  for (int s = 0; s < 2; ++s)
#pragma unroll
    for (int j = 0; j < 4; ++j)
#pragma unroll
      for (int i = 0; i < 8; ++i) acc[s][j][i] = 0.0f;

  const bf16_t* aRow0 = A + (size_t)(mBase + r16) * K;
  const bf16_t* aRow1 = A + (size_t)(mBase + 16 + r16) * K;

#if USE_TDM
  if (wave == 0) { tdm_issue(0, 0); __builtin_amdgcn_s_wait_tensorcnt(0); }
  __syncthreads();
#endif

  // A fragment register prefetch (2 row-sets)
  v8bf aLo0 = *(const v8bf*)(aRow0 + hiKA);
  v8bf aHi0 = *(const v8bf*)(aRow0 + 16 + hiKA);
  v8bf aLo1 = *(const v8bf*)(aRow1 + hiKA);
  v8bf aHi1 = *(const v8bf*)(aRow1 + 16 + hiKA);

  for (int ks = 0; ks < ksteps; ++ks) {
    const int kt = ks << 5;
    const int cur = ks & 1;
#if USE_TDM
    if (wave == 0 && ks + 1 < ksteps) tdm_issue(kt + 32, cur ^ 1);  // prefetch next B tile
#endif
    v16bf aF0, aF1;
#pragma unroll
    for (int i = 0; i < 8; ++i) {
      aF0[i] = aLo0[i]; aF0[i + 8] = aHi0[i];
      aF1[i] = aLo1[i]; aF1[i + 8] = aHi1[i];
    }
    if (ks + 1 < ksteps) {                                          // prefetch next A frags
      aLo0 = *(const v8bf*)(aRow0 + kt + 32 + hiKA);
      aHi0 = *(const v8bf*)(aRow0 + kt + 48 + hiKA);
      aLo1 = *(const v8bf*)(aRow1 + kt + 32 + hiKA);
      aHi1 = *(const v8bf*)(aRow1 + kt + 48 + hiKA);
    }
    // Load all four B fragments first, then issue 8 WMMAs back-to-back.
    v16bf bF[4];
#pragma unroll
    for (int j = 0; j < 4; ++j) {
#if USE_TDM
      const bf16_t* bp = &sB[cur][(j * 16 + r16) * 32 + hiKB];
#else
      const bf16_t* bp = W + (size_t)(nBase + j * 16 + r16) * K + kt + hiKB;
#endif
      v8bf lo = *(const v8bf*)bp;
      v8bf hi = *(const v8bf*)(bp + 16);
#pragma unroll
      for (int i = 0; i < 8; ++i) { bF[j][i] = lo[i]; bF[j][i + 8] = hi[i]; }
    }
#pragma unroll
    for (int j = 0; j < 4; ++j)
      acc[0][j] = __builtin_amdgcn_wmma_f32_16x16x32_bf16(
          false, aF0, false, bF[j], (short)0, acc[0][j], false, false);
#pragma unroll
    for (int j = 0; j < 4; ++j)
      acc[1][j] = __builtin_amdgcn_wmma_f32_16x16x32_bf16(
          false, aF1, false, bF[j], (short)0, acc[1][j], false, false);
#if USE_TDM
    if (wave == 0) __builtin_amdgcn_s_wait_tensorcnt(0);
    __syncthreads();
#endif
  }

#pragma unroll
  for (int s = 0; s < 2; ++s) {
#pragma unroll
    for (int j = 0; j < 4; ++j) {
      const int col = nBase + j * 16 + r16;
      const float bv = bias ? bias[col] : 0.0f;
#pragma unroll
      for (int i = 0; i < 8; ++i) {
        const int row = mBase + s * 16 + hiM + i;
        const size_t idx = (size_t)row * Ncols + col;
        float v = acc[s][j][i] + bv;
        if (mode == 0) {
          outF[idx] = v;
        } else if (mode == 1 || mode == 3) {
          outF[idx] = v + resid[idx];
        } else {
          float g = 0.5f * v * (1.0f + erff(v * 0.70710678118654752f)); // exact GELU
          outB[idx] = (bf16_t)g;
        }
      }
    }
  }
}

// ---------------- per-token SSIM-style q/k statistics ----------------
__global__ __launch_bounds__(256)
void qk_stats_kernel(const float* __restrict__ qkv, const float* __restrict__ c1p,
                     const float* __restrict__ c2p, float* __restrict__ attn_base) {
  __shared__ float red[256];
  const int t = blockIdx.x, tid = threadIdx.x;
  const float* q = qkv + (size_t)t * (3 * CC);
  const float* k = q + CC;
  float q0 = q[tid], q1 = q[tid + 256], k0 = k[tid], k1 = k[tid + 256];
  float sq  = blockReduceSum(q0 + q1, red);
  float sk  = blockReduceSum(k0 + k1, red);
  float sqq = blockReduceSum(q0 * q0 + q1 * q1, red);
  float skk = blockReduceSum(k0 * k0 + k1 * k1, red);
  float sqk = blockReduceSum(q0 * k0 + q1 * k1, red);
  if (tid == 0) {
    const float c1 = c1p[0], c2 = c2p[0];
    const float invC = 1.0f / CC, inv = 1.0f / (CC - 1);
    float muq = sq * invC, muk = sk * invC;
    float sigqk = (sqk - sq * sk * invC) * inv;
    float sigq2 = (sqq - sq * sq * invC) * inv;
    float sigk2 = (skk - sk * sk * invC) * inv;
    float num = (2.f * muq * muk + c1) * (2.f * sigqk + c2);
    float den = (muq * muq + muk * muk + c1) * (sigq2 + sigk2 + c2);
    float r = num / (den + 1e-7f);
    attn_base[t] = r * r;
  }
}

// ---------------- mean over tokens: xm[b,c] ----------------
__global__ __launch_bounds__(256)
void token_mean_kernel(const float* __restrict__ x1, float* __restrict__ xm) {
  int idx = blockIdx.x * 256 + threadIdx.x;      // over B*C
  int b = idx / CC, c = idx - b * CC;
  const float* p = x1 + (size_t)b * NN * CC + c;
  float s = 0.f;
  for (int n = 0; n < NN; ++n) s += p[(size_t)n * CC];
  xm[idx] = s * (1.0f / NN);
}

// ---------------- channel attention MLP: C -> 128 -> C, one block per batch ----------------
__global__ __launch_bounds__(256)
void channel_att_kernel(const float* __restrict__ xm,
                        const float* __restrict__ w1, const float* __restrict__ b1,
                        const float* __restrict__ w2, const float* __restrict__ b2,
                        float* __restrict__ chw) {
  __shared__ float sx[CC];
  __shared__ float sh[CAn];
  const int b = blockIdx.x, tid = threadIdx.x;
  for (int i = tid; i < CC; i += 256) sx[i] = xm[(size_t)b * CC + i];
  __syncthreads();
  if (tid < CAn) {
    float s = b1[tid];
    const float* wr = w1 + (size_t)tid * CC;
    for (int c = 0; c < CC; ++c) s += wr[c] * sx[c];
    sh[tid] = fmaxf(s, 0.f);
  }
  __syncthreads();
  for (int c = tid; c < CC; c += 256) {
    float s = b2[c];
    const float* wr = w2 + (size_t)c * CAn;
    for (int a = 0; a < CAn; ++a) s += wr[a] * sh[a];
    chw[(size_t)b * CC + c] = 1.f / (1.f + expf(-s));
  }
}

// ---------------- ODE (conv->GN->ReLU->conv, Euler dt=1) + sigmoid + token softmax ----------------
__global__ __launch_bounds__(256)
void ode_softmax_kernel(const float* __restrict__ attn_base,
                        const float* __restrict__ w1,
                        const float* __restrict__ gn_g, const float* __restrict__ gn_b,
                        const float* __restrict__ w2, const float* __restrict__ b2,
                        float* __restrict__ wts) {
  __shared__ float sA[NN];
  __shared__ float sH[4][NN];
  __shared__ float red[256];
  const int b = blockIdx.x, tid = threadIdx.x;
  for (int i = tid; i < NN; i += 256) sA[i] = attn_base[(size_t)b * NN + i];
  __syncthreads();
  for (int i = tid; i < NN; i += 256) {
    float xm1 = (i > 0) ? sA[i - 1] : 0.f;
    float x0  = sA[i];
    float xp1 = (i < NN - 1) ? sA[i + 1] : 0.f;
#pragma unroll
    for (int c = 0; c < 4; ++c)
      sH[c][i] = w1[c * 3 + 0] * xm1 + w1[c * 3 + 1] * x0 + w1[c * 3 + 2] * xp1;
  }
  __syncthreads();
  float s0 = 0, ss0 = 0, s1 = 0, ss1 = 0;
  for (int i = tid; i < NN; i += 256) {
    float a = sH[0][i], bb = sH[1][i], c = sH[2][i], d = sH[3][i];
    s0 += a + bb; ss0 += a * a + bb * bb;
    s1 += c + d;  ss1 += c * c + d * d;
  }
  s0 = blockReduceSum(s0, red);  ss0 = blockReduceSum(ss0, red);
  s1 = blockReduceSum(s1, red);  ss1 = blockReduceSum(ss1, red);
  const float invg = 1.0f / (2 * NN);
  float m[2], rs[2];
  m[0] = s0 * invg; m[1] = s1 * invg;
  float v0 = ss0 * invg - m[0] * m[0];
  float v1 = ss1 * invg - m[1] * m[1];
  rs[0] = rsqrtf(v0 + 1e-5f); rs[1] = rsqrtf(v1 + 1e-5f);
  for (int i = tid; i < NN; i += 256) {
#pragma unroll
    for (int c = 0; c < 4; ++c) {
      int g = c >> 1;
      float h = (sH[c][i] - m[g]) * rs[g] * gn_g[c] + gn_b[c];
      sH[c][i] = fmaxf(h, 0.f);
    }
  }
  __syncthreads();
  for (int i = tid; i < NN; i += 256) {
    float o = b2[0];
#pragma unroll
    for (int c = 0; c < 4; ++c) {
      float hm1 = (i > 0) ? sH[c][i - 1] : 0.f;
      float h0  = sH[c][i];
      float hp1 = (i < NN - 1) ? sH[c][i + 1] : 0.f;
      o += w2[c * 3 + 0] * hm1 + w2[c * 3 + 1] * h0 + w2[c * 3 + 2] * hp1;
    }
    float An = sA[i] + o;
    sA[i] = 1.f / (1.f + expf(-An));
  }
  __syncthreads();
  float lmax = -1e30f;
  for (int i = tid; i < NN; i += 256) lmax = fmaxf(lmax, sA[i]);
  float gmax = blockReduceMax(lmax, red);
  float lsum = 0.f;
  for (int i = tid; i < NN; i += 256) { float e = expf(sA[i] - gmax); sA[i] = e; lsum += e; }
  float gsum = blockReduceSum(lsum, red);
  float invs = 1.0f / gsum;
  for (int i = tid; i < NN; i += 256) wts[(size_t)b * NN + i] = sA[i] * invs;
}

// ---------------- out_low = weights * v * ch_w  (-> bf16 for WMMA proj) ----------------
__global__ __launch_bounds__(256)
void out_low_kernel(const float* __restrict__ qkv, const float* __restrict__ wts,
                    const float* __restrict__ chw, bf16_t* __restrict__ out) {
  size_t idx = (size_t)blockIdx.x * 256 + threadIdx.x;   // over B*N*C
  size_t t = idx / CC;
  int c = (int)(idx - t * CC);
  int b = (int)(t / NN);
  float v = qkv[t * (3 * CC) + 2 * CC + c];
  out[idx] = (bf16_t)(wts[t] * v * chw[(size_t)b * CC + c]);
}

// ---------------- launch ----------------
extern "C" void kernel_launch(void* const* d_in, const int* in_sizes, int n_in,
                              void* d_out, int out_size, void* d_ws, size_t ws_size,
                              hipStream_t stream) {
  (void)in_sizes; (void)n_in; (void)out_size; (void)ws_size;
  const float* x      = (const float*)d_in[0];
  const float* ln1_g  = (const float*)d_in[1];
  const float* ln1_b  = (const float*)d_in[2];
  const float* w_qkv  = (const float*)d_in[3];
  const float* c1     = (const float*)d_in[4];
  const float* c2     = (const float*)d_in[5];
  const float* ca_w1  = (const float*)d_in[6];
  const float* ca_b1  = (const float*)d_in[7];
  const float* ca_w2  = (const float*)d_in[8];
  const float* ca_b2  = (const float*)d_in[9];
  const float* ode_w1 = (const float*)d_in[10];
  const float* gn_g   = (const float*)d_in[11];
  const float* gn_b   = (const float*)d_in[12];
  const float* ode_w2 = (const float*)d_in[13];
  const float* ode_b2 = (const float*)d_in[14];
  const float* pl_w   = (const float*)d_in[15];
  const float* pl_b   = (const float*)d_in[16];
  const float* ln2_g  = (const float*)d_in[17];
  const float* ln2_b  = (const float*)d_in[18];
  const float* fc1_w  = (const float*)d_in[19];
  const float* fc1_b  = (const float*)d_in[20];
  const float* fc2_w  = (const float*)d_in[21];
  const float* fc2_b  = (const float*)d_in[22];
  float* out = (float*)d_out;

  char* ws = (char*)d_ws;
  const size_t MB = 1ull << 20;
  float*  x1f   = (float*)(ws + 0);               // 32 MB
  bf16_t* x1b   = (bf16_t*)(ws + 32 * MB);        // 16 MB
  float*  qkv   = (float*)(ws + 48 * MB);         // 96 MB
  float*  attnb = (float*)(ws + 144 * MB);        // 64 KB
  float*  wts   = (float*)(ws + 144 * MB + 64 * 1024);
  float*  xm    = (float*)(ws + 144 * MB + 128 * 1024);
  float*  chw   = (float*)(ws + 144 * MB + 192 * 1024);
  bf16_t* olb   = (bf16_t*)(ws + 145 * MB);       // 16 MB
  float*  xmid  = (float*)(ws + 161 * MB);        // 32 MB
  bf16_t* wqkvb = (bf16_t*)(ws + 193 * MB);       // 1.5 MB
  bf16_t* plwb  = (bf16_t*)(ws + 195 * MB);       // 0.5 MB
  bf16_t* fc1b  = (bf16_t*)(ws + 196 * MB);       // 2 MB
  bf16_t* fc2b  = (bf16_t*)(ws + 198 * MB);       // 2 MB
  bf16_t* hb    = (bf16_t*)(ws + 48 * MB);        // 64 MB, reuses qkv region (qkv dead by then)
  bf16_t* x2b   = (bf16_t*)(ws + 32 * MB);        // reuses x1b region (x1b dead by then)

  // weight fp32 -> bf16
  f2b_kernel<<<(3 * CC * CC + 255) / 256, 256, 0, stream>>>(w_qkv, wqkvb, 3 * CC * CC);
  f2b_kernel<<<(CC * CC + 255) / 256, 256, 0, stream>>>(pl_w, plwb, CC * CC);
  f2b_kernel<<<(HIDD * CC + 255) / 256, 256, 0, stream>>>(fc1_w, fc1b, HIDD * CC);
  f2b_kernel<<<(CC * HIDD + 255) / 256, 256, 0, stream>>>(fc2_w, fc2b, CC * HIDD);

  // LN1
  layernorm_kernel<<<MTOT, 256, 0, stream>>>(x, ln1_g, ln1_b, x1f, x1b);
  // qkv GEMM (WMMA bf16, TDM-staged B, f32 out): grid = (M/256) * (N/64)
  wmma_gemm_kernel<<<(MTOT / 256) * (3 * CC / 64), 256, 0, stream>>>(
      x1b, wqkvb, qkv, nullptr, nullptr, nullptr, MTOT, 3 * CC, CC, 0);
  // SSIM stats -> attn_base
  qk_stats_kernel<<<MTOT, 256, 0, stream>>>(qkv, c1, c2, attnb);
  // channel attention path
  token_mean_kernel<<<(BB * CC) / 256, 256, 0, stream>>>(x1f, xm);
  channel_att_kernel<<<BB, 256, 0, stream>>>(xm, ca_w1, ca_b1, ca_w2, ca_b2, chw);
  // ODE + sigmoid + softmax
  ode_softmax_kernel<<<BB, 256, 0, stream>>>(attnb, ode_w1, gn_g, gn_b, ode_w2, ode_b2, wts);
  // out_low -> bf16
  out_low_kernel<<<(int)(((size_t)MTOT * CC) / 256), 256, 0, stream>>>(qkv, wts, chw, olb);
  // proj GEMM + bias + residual -> x_mid
  wmma_gemm_kernel<<<(MTOT / 256) * (CC / 64), 256, 0, stream>>>(
      olb, plwb, xmid, nullptr, pl_b, x, MTOT, CC, CC, 1);
  // LN2 -> bf16
  layernorm_kernel<<<MTOT, 256, 0, stream>>>(xmid, ln2_g, ln2_b, nullptr, x2b);
  // fc1 GEMM + GELU -> bf16 h
  wmma_gemm_kernel<<<(MTOT / 256) * (HIDD / 64), 256, 0, stream>>>(
      x2b, fc1b, nullptr, hb, fc1_b, nullptr, MTOT, HIDD, CC, 2);
  // fc2 GEMM + bias + residual -> out
  wmma_gemm_kernel<<<(MTOT / 256) * (CC / 64), 256, 0, stream>>>(
      hb, fc2b, out, nullptr, fc2_b, xmid, MTOT, CC, HIDD, 3);
}
